// DgcnnModel_35407710388660
// MI455X (gfx1250) — matrix operand order repaired and checked by reading, not compiled, not run on hardware
//
#include <hip/hip_runtime.h>
#include <stddef.h>
#include <stdint.h>

// ---------------- constants ----------------
#define BB 8
#define NN 4096
#define KNN 20
#define HCAT_W 512
#define NPTS (BB * NN)
#define BN_COUNT 655360.0f   // B*N*K = 8*4096*20

typedef __attribute__((ext_vector_type(16))) __bf16 v16bf;
typedef __attribute__((ext_vector_type(8)))  float  v8f;

union ABf { unsigned int u[8]; unsigned short h[16]; v16bf v; };

#if __has_builtin(__builtin_amdgcn_tensor_load_to_lds)
#define HAVE_TDM 1
typedef __attribute__((ext_vector_type(4))) unsigned int tdm_u4;
typedef __attribute__((ext_vector_type(8))) int tdm_i8;
typedef __attribute__((ext_vector_type(4))) int tdm_i4;
#else
#define HAVE_TDM 0
#endif

// float -> bf16 (RNE), stored as raw u16 bits
__device__ __forceinline__ unsigned short f2bf(float x) {
  unsigned int u = __float_as_uint(x);
  unsigned int r = u + 0x7fffu + ((u >> 16) & 1u);
  return (unsigned short)(r >> 16);
}

__device__ __forceinline__ v8f wmma_bf16(v16bf a, v16bf b, v8f c) {
  return __builtin_amdgcn_wmma_f32_16x16x32_bf16(false, a, false, b, (short)0, c, false, false);
}

// A fragment: 16x32 bf16, row M = lane&15; K contiguous: k = 16*half + s
__device__ __forceinline__ v16bf load_a_frag(const unsigned short* A, int lda,
                                             int row0, int k0, int lane) {
  ABf f;
  int m = lane & 15, half = lane >> 4;
  const unsigned short* rp = A + (size_t)(row0 + m) * lda + k0;
#pragma unroll
  for (int v = 0; v < 8; ++v) {
    int kb = half * 16 + 2 * v;
    f.u[v] = *(const unsigned int*)(rp + kb);
  }
  return f.v;
}

// B fragment 32x16 where B[k][n] = M[col0+n][k] for a row-major matrix M[ncols][ldh].
// Used both for the gram (candidate points as columns) and for *transposed*
// weights Wt[Dout][Kpad] (so every lane reads 16 contiguous ushorts -> b128s).
__device__ __forceinline__ v16bf load_b_rows(const unsigned short* H, int ldh,
                                             int col0, int k0, int lane) {
  ABf f;
  int n = lane & 15, half = lane >> 4;
  const unsigned short* cp = H + (size_t)(col0 + n) * ldh + k0 + half * 16;
#pragma unroll
  for (int v = 0; v < 8; ++v) f.u[v] = *(const unsigned int*)(cp + 2 * v);
  return f.v;
}

__device__ __forceinline__ void atomicMaxF(float* addr, float v) {
  if (v >= 0.f) atomicMax((int*)addr, __float_as_int(v));
  else          atomicMin((unsigned int*)addr, __float_as_uint(v));
}

__device__ __forceinline__ void wait_tensor0() {
#if __has_builtin(__builtin_amdgcn_s_wait_tensorcnt)
  __builtin_amdgcn_s_wait_tensorcnt(0);
#else
  asm volatile("s_wait_tensorcnt 0x0" ::: "memory");
#endif
}

#if HAVE_TDM
// TDM: DMA `rows` consecutive rows of `rowlen` bf16 elements (row-major,
// stride == rowlen) from global into LDS at byte offset ldsOff.
// D# per CDNA5 ISA 8.3/8.4: group0 = {count, lds_addr, global_addr, type=2},
// group1 = {data_size=2B, tensor dims, tile dims, dim0 stride}.
// This toolchain exposes the 6-arg builtin:
//   (uint32x4 g0, int32x8 g1, int32x4 g2, int32x4 g3, int32x8 extra, i32 cpol)
__device__ __forceinline__ void tdm_load_rows(const unsigned short* gsrc,
                                              unsigned ldsOff, int rows, int rowlen) {
  unsigned long long ga = (unsigned long long)(uintptr_t)gsrc;
  tdm_u4 g0;
  g0[0] = 1u;                                            // count=1 (valid user D#)
  g0[1] = ldsOff;                                        // lds_addr (bytes)
  g0[2] = (unsigned)ga;                                  // global_addr[31:0]
  g0[3] = (unsigned)((ga >> 32) & 0x01ffffffull) | 0x80000000u; // addr[56:32] | type=2
  tdm_i8 g1;
  g1[0] = 0x00010000;                                    // data_size=1 -> 2 bytes/elem
  g1[1] = (int)((rowlen & 0xffff) << 16);                // tensor_dim0[15:0]
  g1[2] = (int)((((unsigned)rowlen) >> 16) | ((rows & 0xffff) << 16)); // dim0 hi | tensor_dim1 lo
  g1[3] = (int)((((unsigned)rows) >> 16) | ((rowlen & 0xffff) << 16)); // dim1 hi | tile_dim0
  g1[4] = (int)(rows & 0xffff);                          // tile_dim1 (tile_dim2 = 0)
  g1[5] = rowlen;                                        // tensor_dim0_stride[31:0]
  g1[6] = 0;                                             // stride0 hi | stride1 lo
  g1[7] = 0;
  tdm_i4 gz; gz[0] = 0; gz[1] = 0; gz[2] = 0; gz[3] = 0;
  tdm_i8 gz8;
#pragma unroll
  for (int i = 0; i < 8; ++i) gz8[i] = 0;
  __builtin_amdgcn_tensor_load_to_lds(g0, g1, gz, gz, gz8, 0);
}
#endif

// ---------------- elementwise / prep kernels ----------------

__global__ void zerof_kernel(float* p, int n) {
  int i = blockIdx.x * blockDim.x + threadIdx.x;
  if (i < n) p[i] = 0.f;
}

// f32 weights W[Din][Dout] -> TRANSPOSED padded bf16 Wt[Dout][Kpad]
// (out-feature-major, so B-fragment loads are contiguous per lane)
__global__ void cvt_w_kernel(const float* W, int Din, int Dout, int Kpad,
                             unsigned short* Wt) {
  int i = blockIdx.x * blockDim.x + threadIdx.x;
  if (i >= Dout * Kpad) return;
  int c = i / Kpad, k = i - c * Kpad;
  Wt[i] = f2bf(k < Din ? W[(size_t)k * Dout + c] : 0.f);
}

__global__ void addvec_kernel(const float* a, const float* b, float* o, int n) {
  int i = blockIdx.x * blockDim.x + threadIdx.x;
  if (i < n) o[i] = a[i] + b[i];
}

// convert per-point features (strided f32) -> compact padded bf16, plus sq-norms
__global__ void cvt_h_kernel(const float* src, int srcStride, int srcOff,
                             int Din, int Kpad, unsigned short* hb, float* d2) {
  int p = blockIdx.x * blockDim.x + threadIdx.x;
  if (p >= NPTS) return;
  const float* r = src + (size_t)p * srcStride + srcOff;
  unsigned short* o = hb + (size_t)p * Kpad;
  float s = 0.f;
  for (int d = 0; d < Kpad; ++d) {
    float v = (d < Din) ? r[d] : 0.f;
    s += v * v;
    o[d] = f2bf(v);
  }
  if (d2) d2[p] = s;
}

// ---------------- kNN: WMMA gram + fused top-20 ----------------
// block = 128 threads (4 waves), handles 16 query rows of one batch.
// KP (padded feature dim) is a template parameter so the A-fragment array is
// statically indexed (no v_movrels) and the K-loop fully unrolls.
// Candidate 64-row blocks are double-buffered into LDS via the TDM.
template <int KP>
__global__ void knn_kernel(const unsigned short* __restrict__ hb,
                           const float* __restrict__ d2, int* __restrict__ idxOut) {
  constexpr int NK = KP / 32;
  __shared__ float dbuf[16][65];
#if HAVE_TDM
  __shared__ __align__(16) unsigned short stage[2][64 * KP];
#endif
  int blk = blockIdx.x;
  int b = blk / (NN / 16);
  int m0 = (blk % (NN / 16)) * 16;
  int tid = threadIdx.x;
  int wave = tid >> 5, lane = tid & 31;
  int n = lane & 15, half = lane >> 4;
  const unsigned short* H = hb + (size_t)b * NN * KP;
  const float* D2 = d2 + (size_t)b * NN;

  float rowd2[8];
#pragma unroll
  for (int v = 0; v < 8; ++v) rowd2[v] = D2[m0 + v + half * 8];

  // hoist loop-invariant A fragments (query rows) into registers
  v16bf afr[NK];
#pragma unroll
  for (int t = 0; t < NK; ++t) afr[t] = load_a_frag(H, KP, m0, t * 32, lane);

  float tk_d[KNN]; int tk_i[KNN];
#pragma unroll
  for (int i = 0; i < KNN; ++i) { tk_d[i] = 3.0e38f; tk_i[i] = 0; }
  float worst = 3.0e38f; int wpos = 0;

#if HAVE_TDM
  unsigned lds0 = (unsigned)(uintptr_t)&stage[0][0];
  unsigned lds1 = (unsigned)(uintptr_t)&stage[1][0];
  if (wave == 0) {
    tdm_load_rows(H, lds0, 64, KP);
    wait_tensor0();
  }
  __syncthreads();
#endif

  for (int chunk = 0; chunk < NN / 64; ++chunk) {
#if HAVE_TDM
    const unsigned short* Bbase = &stage[chunk & 1][0];
    int c0 = wave * 16;  // column offset within staged chunk
    if (wave == 0 && chunk + 1 < NN / 64)
      tdm_load_rows(H + (size_t)(chunk + 1) * 64 * KP,
                    (chunk & 1) ? lds0 : lds1, 64, KP);
#else
    const unsigned short* Bbase = H;
    int c0 = chunk * 64 + wave * 16;
    if (chunk + 1 < NN / 64)
      __builtin_prefetch(H + (size_t)(c0 + 64) * KP, 0, 1);
#endif
    v8f acc = {0.f, 0.f, 0.f, 0.f, 0.f, 0.f, 0.f, 0.f};
#pragma unroll
    for (int t = 0; t < NK; ++t)
      acc = wmma_bf16(afr[t], load_b_rows(Bbase, KP, c0, t * 32, lane), acc);

    float cd2 = D2[chunk * 64 + wave * 16 + n];
#pragma unroll
    for (int v = 0; v < 8; ++v) {
      int m = v + half * 8;
      dbuf[m][wave * 16 + n] = rowd2[v] + cd2 - 2.0f * acc[v];
    }
    __syncthreads();
    if (tid < 16) {
#pragma unroll 1
      for (int j = 0; j < 64; ++j) {
        float d = dbuf[tid][j];
        if (d < worst) {
          tk_d[wpos] = d; tk_i[wpos] = chunk * 64 + j;
          worst = tk_d[0]; wpos = 0;
#pragma unroll
          for (int i = 1; i < KNN; ++i)
            if (tk_d[i] > worst) { worst = tk_d[i]; wpos = i; }
        }
      }
    }
#if HAVE_TDM
    if (wave == 0 && chunk + 1 < NN / 64) wait_tensor0();
#endif
    __syncthreads();
  }
  if (tid < 16) {
    int* op = idxOut + ((size_t)b * NN + m0 + tid) * KNN;
#pragma unroll
    for (int i = 0; i < KNN; ++i) op[i] = tk_i[i];
  }
}

// ---------------- generic bf16 GEMM  C[M,N] = A[M,Kpad] x Wt^T + bias ----------------
// A[M,Kpad] bf16 row-major; Wt[N][Kpad] bf16 (transposed weights).
// block = 128 (4 waves), grid = (M/16, N/64); each wave one 16x16 tile
__global__ void gemm_bf16_kernel(const unsigned short* A, int lda,
                                 const unsigned short* Wt,
                                 const float* bias, float* C, int ldc, int Kpad) {
  int wave = threadIdx.x >> 5, lane = threadIdx.x & 31;
  int m0 = blockIdx.x * 16;
  int n0 = blockIdx.y * 64 + wave * 16;
  v8f acc = {0.f, 0.f, 0.f, 0.f, 0.f, 0.f, 0.f, 0.f};
  for (int ks = 0; ks < Kpad; ks += 32) {
    v16bf a = load_a_frag(A, lda, m0, ks, lane);
    v16bf b = load_b_rows(Wt, Kpad, n0, ks, lane);
    acc = wmma_bf16(a, b, acc);
  }
  int n = lane & 15, half = lane >> 4;
  float bv = bias ? bias[n0 + n] : 0.f;
#pragma unroll
  for (int v = 0; v < 8; ++v)
    C[(size_t)(m0 + v + half * 8) * ldc + n0 + n] = acc[v] + bv;
}

// ---------------- EdgeConv: theta(diff) + phix; PASS1 stats, PASS2 BN+max+leaky ----------------
template <int PASS>
__global__ void edgeconv_kernel(const float* hin, int hinStride, int hinOff,
                                int Din, int Kpad,
                                const unsigned short* thWt, int Dout,
                                const float* phix, const int* knnidx,
                                const float* gamma, const float* beta,
                                float* stat_sum, float* stat_sq,
                                const float* mu, const float* rsig,
                                float* hcat_out, int outOff) {
  __shared__ __align__(16) unsigned short diff[32 * 128];
  __shared__ float sbuf[256];
  __shared__ float qbuf[256];
  int p = blockIdx.x;
  int b = p / NN;
  const float* hrow = hin + (size_t)p * hinStride + hinOff;
  const int* nb = knnidx + (size_t)p * KNN;

  for (int e = threadIdx.x; e < 32 * Kpad; e += 256) {
    int r = e / Kpad, c = e - r * Kpad;
    float v = 0.f;
    if (r < KNN && c < Din) {
      int j = nb[r];
      v = hin[((size_t)b * NN + j) * hinStride + hinOff + c] - hrow[c];
    }
    diff[e] = f2bf(v);
  }
  for (int f = threadIdx.x; f < Dout; f += 256) {
    sbuf[f] = (PASS == 1) ? 0.f : -3.0e38f;
    if (PASS == 1) qbuf[f] = 0.f;
  }
  __syncthreads();

  int wave = threadIdx.x >> 5, lane = threadIdx.x & 31;
  int n = lane & 15, half = lane >> 4;
  int ntiles = 2 * (Dout >> 4);
  for (int t = wave; t < ntiles; t += 8) {
    int mt = t & 1, ft = t >> 1;
    v8f acc = {0.f, 0.f, 0.f, 0.f, 0.f, 0.f, 0.f, 0.f};
    for (int ks = 0; ks < Kpad; ks += 32) {
      v16bf a  = load_a_frag(diff, Kpad, mt * 16, ks, lane);
      v16bf bw = load_b_rows(thWt, Kpad, ft * 16, ks, lane);
      acc = wmma_bf16(a, bw, acc);
    }
    int f = ft * 16 + n;
    float px = phix[(size_t)p * Dout + f];
    if (PASS == 1) {
      float s = 0.f, q = 0.f;
#pragma unroll
      for (int v = 0; v < 8; ++v) {
        int r = mt * 16 + v + half * 8;
        if (r < KNN) { float e = acc[v] + px; s += e; q += e * e; }
      }
      atomicAdd(&sbuf[f], s);
      atomicAdd(&qbuf[f], q);
    } else {
      float g  = gamma[f] * rsig[f];
      float bt = beta[f] - mu[f] * g;
      float mx = -3.0e38f;
#pragma unroll
      for (int v = 0; v < 8; ++v) {
        int r = mt * 16 + v + half * 8;
        if (r < KNN) { float e = (acc[v] + px) * g + bt; mx = fmaxf(mx, e); }
      }
      atomicMaxF(&sbuf[f], mx);
    }
  }
  __syncthreads();
  if (PASS == 1) {
    for (int f = threadIdx.x; f < Dout; f += 256) {
      atomicAdd(&stat_sum[f], sbuf[f]);
      atomicAdd(&stat_sq[f], qbuf[f]);
    }
  } else {
    for (int f = threadIdx.x; f < Dout; f += 256) {
      float hv = sbuf[f];
      hcat_out[(size_t)p * HCAT_W + outOff + f] = hv < 0.f ? hv * 0.2f : hv;
    }
  }
}

__global__ void bn_finalize_kernel(const float* s, const float* q,
                                   float* mu, float* rsig, int Dout) {
  int f = blockIdx.x * blockDim.x + threadIdx.x;
  if (f >= Dout) return;
  float m = s[f] / BN_COUNT;
  float var = q[f] / BN_COUNT - m * m;
  mu[f] = m;
  rsig[f] = rsqrtf(var + 1e-5f);
}

// ---------------- proj GEMM with fused max/avg pooling ----------------
__global__ void pool_init_kernel(float* psum, float* pmax) {
  int i = blockIdx.x * blockDim.x + threadIdx.x;
  if (i < BB * 512) { psum[i] = 0.f; pmax[i] = -3.0e38f; }
}

__global__ void proj_pool_kernel(const unsigned short* A, const unsigned short* Wt,
                                 float* psum, float* pmax) {
  int wave = threadIdx.x >> 5, lane = threadIdx.x & 31;
  int m0 = blockIdx.x * 16;
  int b = m0 / NN;
  int n0 = blockIdx.y * 64 + wave * 16;
  v8f acc = {0.f, 0.f, 0.f, 0.f, 0.f, 0.f, 0.f, 0.f};
  for (int ks = 0; ks < 512; ks += 32) {
    v16bf a  = load_a_frag(A, 512, m0, ks, lane);
    v16bf bb = load_b_rows(Wt, 512, n0, ks, lane);
    acc = wmma_bf16(a, bb, acc);
  }
  float s = 0.f, mx = -3.0e38f;
#pragma unroll
  for (int v = 0; v < 8; ++v) { s += acc[v]; mx = fmaxf(mx, acc[v]); }
  s  += __shfl_xor(s, 16, 32);
  mx  = fmaxf(mx, __shfl_xor(mx, 16, 32));
  if (lane < 16) {
    atomicAdd(&psum[b * 512 + n0 + lane], s);
    atomicMaxF(&pmax[b * 512 + n0 + lane], mx);
  }
}

__global__ void pool_fin_kernel(const float* psum, const float* pmax,
                                const float* projb, float* hpool) {
  int i = blockIdx.x * blockDim.x + threadIdx.x;
  if (i >= BB * 512) return;
  int b = i / 512, f = i - b * 512;
  hpool[(size_t)b * 1024 + f]       = pmax[i] + projb[f];
  hpool[(size_t)b * 1024 + 512 + f] = psum[i] / (float)NN + projb[f];
}

// ---------------- tiny MLP tail (exact f32) ----------------
__global__ void mlp_kernel(const float* in, int K, const float* W,
                           const float* bias, float* out, int N, int relu) {
  int i = blockIdx.x * blockDim.x + threadIdx.x;
  if (i >= BB * N) return;
  int r = i / N, c = i - r * N;
  float s = bias[c];
  for (int k = 0; k < K; ++k) s += in[(size_t)r * K + k] * W[(size_t)k * N + c];
  if (relu && s < 0.f) s *= 0.2f;
  out[i] = s;
}

// ---------------- host orchestration ----------------

// workspace layout (bytes); total ~137.8 MB
static constexpr size_t O_HCAT  = 0;                          // 8*4096*512*4
static constexpr size_t O_HB    = O_HCAT + 67108864;          // 32768*512*2
static constexpr size_t O_D2    = O_HB + 33554432;            // 32768*4
static constexpr size_t O_IDX   = O_D2 + 131072;              // 32768*20*4
static constexpr size_t O_PHIX  = O_IDX + 2621440;            // 32768*256*4
static constexpr size_t O_WB    = O_PHIX + 33554432;          // 356352*2
static constexpr size_t O_BIAS  = O_WB + 712704;              // 512*4
static constexpr size_t O_STAT  = O_BIAS + 2048;              // sum256+sq256+mu256+rsig256
static constexpr size_t O_POOL  = O_STAT + 4096;              // psum 4096f + pmax 4096f
static constexpr size_t O_HPOOL = O_POOL + 32768;             // 8*1024*4
static constexpr size_t O_T1    = O_HPOOL + 32768;            // 8*512*4
static constexpr size_t O_T2    = O_T1 + 16384;               // 8*256*4

extern "C" void kernel_launch(void* const* d_in, const int* in_sizes, int n_in,
                              void* d_out, int out_size, void* d_ws, size_t ws_size,
                              hipStream_t stream) {
  (void)n_in; (void)out_size; (void)ws_size;

  static const int Din_[4]     = {3, 64, 64, 128};
  static const int Kpad_[4]    = {32, 64, 64, 128};
  static const int Dout_[4]    = {64, 64, 128, 256};
  static const int hOffIn_[4]  = {0, 0, 64, 128};
  static const int hOffOut_[4] = {0, 64, 128, 256};
  static const int thOff_[4]   = {0, 2048, 6144, 14336};
  const int PHI_BASE  = 47104;
  const int PROJ_BASE = 94208;

  const float *xPtr, *thW[4], *thB[4], *phW[4], *phB[4], *gam[4], *bet[4];
  const float *projW, *projB, *embW[2], *embB[2], *outW, *outB;

  if (in_sizes[0] == BB * NN * 3) {
    // insertion order: x, then layers(theta_W,theta_b,phi_W,phi_b,gamma,beta)x4,
    // proj_W, proj_b, embs, out_W, out_b
    int i = 0;
    xPtr = (const float*)d_in[i++];
    for (int l = 0; l < 4; ++l) {
      thW[l] = (const float*)d_in[i++]; thB[l] = (const float*)d_in[i++];
      phW[l] = (const float*)d_in[i++]; phB[l] = (const float*)d_in[i++];
      gam[l] = (const float*)d_in[i++]; bet[l] = (const float*)d_in[i++];
    }
    projW = (const float*)d_in[i++]; projB = (const float*)d_in[i++];
    embW[0] = (const float*)d_in[i++]; embB[0] = (const float*)d_in[i++];
    embW[1] = (const float*)d_in[i++]; embB[1] = (const float*)d_in[i++];
    outW = (const float*)d_in[i++]; outB = (const float*)d_in[i++];
  } else {
    // jax tree_flatten sorted-key order:
    // embs(e0W,e0b,e1W,e1b), layers(beta,gamma,phi_W,phi_b,theta_W,theta_b)x4,
    // out_W, out_b, proj_W, proj_b, x
    int i = 0;
    embW[0] = (const float*)d_in[i++]; embB[0] = (const float*)d_in[i++];
    embW[1] = (const float*)d_in[i++]; embB[1] = (const float*)d_in[i++];
    for (int l = 0; l < 4; ++l) {
      bet[l] = (const float*)d_in[i++]; gam[l] = (const float*)d_in[i++];
      phW[l] = (const float*)d_in[i++]; phB[l] = (const float*)d_in[i++];
      thW[l] = (const float*)d_in[i++]; thB[l] = (const float*)d_in[i++];
    }
    outW = (const float*)d_in[i++]; outB = (const float*)d_in[i++];
    projW = (const float*)d_in[i++]; projB = (const float*)d_in[i++];
    xPtr = (const float*)d_in[i++];
  }

  char* ws = (char*)d_ws;
  float*          hcat  = (float*)(ws + O_HCAT);
  unsigned short* hb    = (unsigned short*)(ws + O_HB);
  float*          d2    = (float*)(ws + O_D2);
  int*            knni  = (int*)(ws + O_IDX);
  float*          phix  = (float*)(ws + O_PHIX);
  unsigned short* wb    = (unsigned short*)(ws + O_WB);
  float*          biasc = (float*)(ws + O_BIAS);
  float*          ssum  = (float*)(ws + O_STAT);
  float*          ssq   = ssum + 256;
  float*          mu    = ssum + 512;
  float*          rsig  = ssum + 768;
  float*          psum  = (float*)(ws + O_POOL);
  float*          pmax  = psum + BB * 512;
  float*          hpool = (float*)(ws + O_HPOOL);
  float*          t1    = (float*)(ws + O_T1);
  float*          t2    = (float*)(ws + O_T2);
  float*          outP  = (float*)d_out;

  // ---- weight conversions (f32 -> transposed padded bf16) ----
  for (int l = 0; l < 4; ++l) {
    int ne = Kpad_[l] * Dout_[l];
    cvt_w_kernel<<<(ne + 255) / 256, 256, 0, stream>>>(thW[l], Din_[l], Dout_[l], Kpad_[l], wb + thOff_[l]);
    cvt_w_kernel<<<(ne + 255) / 256, 256, 0, stream>>>(phW[l], Din_[l], Dout_[l], Kpad_[l], wb + PHI_BASE + thOff_[l]);
    addvec_kernel<<<1, Dout_[l], 0, stream>>>(thB[l], phB[l], biasc + hOffOut_[l], Dout_[l]);
  }
  cvt_w_kernel<<<(512 * 512 + 255) / 256, 256, 0, stream>>>(projW, 512, 512, 512, wb + PROJ_BASE);

  // ---- EdgeConv layers ----
  for (int l = 0; l < 4; ++l) {
    const float* hin = (l == 0) ? xPtr : hcat;
    int hst  = (l == 0) ? 3 : HCAT_W;
    int hoff = (l == 0) ? 0 : hOffIn_[l];
    int Din = Din_[l], Kp = Kpad_[l], Do = Dout_[l];

    cvt_h_kernel<<<NPTS / 256, 256, 0, stream>>>(hin, hst, hoff, Din, Kp, hb, d2);
    if (Kp == 32)       knn_kernel<32><<<NPTS / 16, 128, 0, stream>>>(hb, d2, knni);
    else if (Kp == 64)  knn_kernel<64><<<NPTS / 16, 128, 0, stream>>>(hb, d2, knni);
    else                knn_kernel<128><<<NPTS / 16, 128, 0, stream>>>(hb, d2, knni);

    dim3 gg(NPTS / 16, Do >= 64 ? Do / 64 : 1);
    gemm_bf16_kernel<<<gg, 128, 0, stream>>>(hb, Kp, wb + PHI_BASE + thOff_[l],
                                             biasc + hOffOut_[l], phix, Do, Kp);

    zerof_kernel<<<2, 256, 0, stream>>>(ssum, 512);
    edgeconv_kernel<1><<<NPTS, 256, 0, stream>>>(hin, hst, hoff, Din, Kp,
                                                 wb + thOff_[l], Do, phix, knni,
                                                 gam[l], bet[l], ssum, ssq, mu, rsig,
                                                 hcat, hOffOut_[l]);
    bn_finalize_kernel<<<1, 256, 0, stream>>>(ssum, ssq, mu, rsig, Do);
    edgeconv_kernel<2><<<NPTS, 256, 0, stream>>>(hin, hst, hoff, Din, Kp,
                                                 wb + thOff_[l], Do, phix, knni,
                                                 gam[l], bet[l], ssum, ssq, mu, rsig,
                                                 hcat, hOffOut_[l]);
  }

  // ---- proj + fused global pooling ----
  cvt_h_kernel<<<NPTS / 256, 256, 0, stream>>>(hcat, HCAT_W, 0, 512, 512, hb, nullptr);
  pool_init_kernel<<<(BB * 512 + 255) / 256, 256, 0, stream>>>(psum, pmax);
  proj_pool_kernel<<<dim3(NPTS / 16, 8), 128, 0, stream>>>(hb, wb + PROJ_BASE, psum, pmax);
  pool_fin_kernel<<<(BB * 512 + 255) / 256, 256, 0, stream>>>(psum, pmax, projB, hpool);

  // ---- MLP head (exact f32) ----
  mlp_kernel<<<(BB * 512 + 255) / 256, 256, 0, stream>>>(hpool, 1024, embW[0], embB[0], t1, 512, 1);
  mlp_kernel<<<(BB * 256 + 255) / 256, 256, 0, stream>>>(t1, 512, embW[1], embB[1], t2, 256, 1);
  mlp_kernel<<<(BB * 40 + 255) / 256, 256, 0, stream>>>(t2, 256, outW, outB, outP, 40, 0);
}
